// MuralDecoderModule_30528627540337
// MI455X (gfx1250) — matrix operand order repaired and checked by reading, not compiled, not run on hardware
//
#include <hip/hip_runtime.h>
#include <hip/hip_bf16.h>

// ---------------------------------------------------------------------------
// Types for CDNA5 WMMA
// ---------------------------------------------------------------------------
typedef __attribute__((ext_vector_type(16))) __bf16        v16bf;
typedef __attribute__((ext_vector_type(8)))  float          v8f;
typedef __attribute__((ext_vector_type(4)))  unsigned int   v4u;
typedef __attribute__((ext_vector_type(16))) unsigned short v16us;

union BV {
    v16bf bf;
    v16us us;
    v4u   u4[2];
};

static __device__ __forceinline__ v8f wmma_bf16(v16bf a, v16bf b, v8f c) {
    // D = A(16x32) * B(32x16) + C, f32 accum
    return __builtin_amdgcn_wmma_f32_16x16x32_bf16(
        /*neg_a=*/false, a, /*neg_b=*/false, b,
        /*c_mod=*/(short)0, c, /*reuse_a=*/false, /*reuse_b=*/false);
}

static __device__ __forceinline__ unsigned short f2bf(float f) {
    unsigned u = __float_as_uint(f);
    u += 0x7FFFu + ((u >> 16) & 1u);   // round-to-nearest-even
    return (unsigned short)(u >> 16);
}

// Problem constants
#define NB    8
#define NC    256
#define NH    128
#define NW    128
#define NTOK  131072L          // 8*128*128
#define NWIN  2048             // 8*16*16
#define HEADS 8
#define HD    32               // head dim

// ---------------------------------------------------------------------------
// Kernel 0: NCHW -> NHWC transpose (fp32), tiled through LDS
// ---------------------------------------------------------------------------
__global__ __launch_bounds__(256) void k_transpose(const float* __restrict__ x,
                                                   float* __restrict__ xc) {
    __shared__ float tile[32][33];
    int b  = blockIdx.z;
    int c0 = blockIdx.y * 32;      // channel tile
    int p0 = blockIdx.x * 32;      // pixel tile (h*W+w)
    int tx = threadIdx.x, ty = threadIdx.y;   // 32 x 8
#pragma unroll
    for (int k = 0; k < 4; ++k)
        tile[ty + k * 8][tx] =
            x[((long)b * NC + c0 + ty + k * 8) * (NH * NW) + p0 + tx];
    __syncthreads();
#pragma unroll
    for (int k = 0; k < 4; ++k)
        xc[((long)b * (NH * NW) + p0 + ty + k * 8) * NC + c0 + tx] =
            tile[tx][ty + k * 8];
}

// ---------------------------------------------------------------------------
// Weight pre-swizzle into WMMA-B-operand layout (bf16).
// B k order used here (contiguous, same as A): lane l, half h -> K = kc*32 + (l>>4)*16 + h, N = nt*16 + (l&15)
// dst[((kc*NT + nt)*32 + lane)*16 + h]
// ---------------------------------------------------------------------------
__global__ __launch_bounds__(256) void k_prep_w(const float* __restrict__ W,
                                                unsigned short* __restrict__ dst,
                                                int K, int N) {
    int NT    = N >> 4;
    int total = (K >> 5) * NT * 32;
    int tid   = blockIdx.x * 256 + threadIdx.x;
    if (tid >= total) return;
    int lane = tid & 31;
    int nt   = (tid >> 5) % NT;
    int kc   = tid / (32 * NT);
    int n    = nt * 16 + (lane & 15);
    int kb   = kc * 32 + (lane >> 4) * 16;
    unsigned short* d = dst + ((long)(kc * NT + nt) * 32 + lane) * 16;
#pragma unroll
    for (int h = 0; h < 16; ++h)
        d[h] = f2bf(W[(long)(kb + h) * N + n]);
}

// ---------------------------------------------------------------------------
// LayerNorm: one token (C=256) per wave; wave32 shuffle reduction.
// window_shuffle: 0 = token-major out, 1 = window-partitioned out.
// ---------------------------------------------------------------------------
__global__ __launch_bounds__(256) void k_layernorm(const float* __restrict__ x,
                                                   const float* __restrict__ sc,
                                                   const float* __restrict__ bi,
                                                   unsigned short* __restrict__ y,
                                                   int window_shuffle) {
    long tk   = (long)blockIdx.x * 8 + (threadIdx.x >> 5);
    int  lane = threadIdx.x & 31;
    const float* row = x + tk * NC;
    float v[8], s = 0.f;
#pragma unroll
    for (int i = 0; i < 8; ++i) { v[i] = row[lane + i * 32]; s += v[i]; }
#pragma unroll
    for (int m = 1; m < 32; m <<= 1) s += __shfl_xor(s, m, 32);
    float mu = s * (1.0f / 256.0f);
    float s2 = 0.f;
#pragma unroll
    for (int i = 0; i < 8; ++i) { float d = v[i] - mu; s2 += d * d; }
#pragma unroll
    for (int m = 1; m < 32; m <<= 1) s2 += __shfl_xor(s2, m, 32);
    float rs = rsqrtf(s2 * (1.0f / 256.0f) + 1e-5f);

    long drow = tk;
    if (window_shuffle) {
        int b = (int)(tk >> 14), h = (int)(tk >> 7) & 127, w = (int)tk & 127;
        int win = b * 256 + (h >> 3) * 16 + (w >> 3);
        int t   = (h & 7) * 8 + (w & 7);
        drow = (long)win * 64 + t;
    }
    unsigned short* o = y + drow * NC;
#pragma unroll
    for (int i = 0; i < 8; ++i) {
        int c = lane + i * 32;
        o[c] = f2bf((v[i] - mu) * rs * sc[c] + bi[c]);
    }
}

// ---------------------------------------------------------------------------
// Generic WMMA GEMM core: one wave computes a 32(M) x 64(N) tile.
// A: bf16 activations, row-major [M][lda]; W: pre-swizzled bf16 weights.
// A k order used here (contiguous, same as B): lane l, half h -> M=l&15, K=(l>>4)*16 + h
// ---------------------------------------------------------------------------
static __device__ __forceinline__ void gemm_core(const unsigned short* __restrict__ A,
                                                 int lda,
                                                 const unsigned short* __restrict__ W,
                                                 int NT, int nchunks,
                                                 long row0, int nt0,
                                                 v8f c[2][4]) {
    const int lane = threadIdx.x & 31;
    const int g = lane >> 4, ncol = lane & 15;
    for (int kc = 0; kc < nchunks; ++kc) {
        BV a0, a1;
        const unsigned short* p = A + (row0 + ncol) * (long)lda + kc * 32 + g * 16;
        a0.u4[0] = *(const v4u*)p;          a0.u4[1] = *(const v4u*)(p + 8);
        p += 16L * lda;
        a1.u4[0] = *(const v4u*)p;          a1.u4[1] = *(const v4u*)(p + 8);
#pragma unroll
        for (int nt = 0; nt < 4; ++nt) {
            BV b;
            const unsigned short* wp =
                W + ((long)(kc * NT + nt0 + nt) * 32 + lane) * 16;
            b.u4[0] = *(const v4u*)wp;      b.u4[1] = *(const v4u*)(wp + 8);
            c[0][nt] = wmma_bf16(a0.bf, b.bf, c[0][nt]);
            c[1][nt] = wmma_bf16(a1.bf, b.bf, c[1][nt]);
        }
    }
}
// C/D f32 layout: vgpr j, lane l -> M = j + (l>>4)*8, N = l&15

// --- GEMM 1: qkv = xn_win @ w_qkv + b_qkv  (q columns pre-scaled), bf16 out
__global__ __launch_bounds__(256) void k_gemm_qkv(const unsigned short* __restrict__ A,
                                                  const unsigned short* __restrict__ W,
                                                  const float* __restrict__ bias,
                                                  unsigned short* __restrict__ out) {
    const int lane = threadIdx.x & 31, g = lane >> 4, ncol = lane & 15;
    long row0 = ((long)blockIdx.x * 8 + (threadIdx.x >> 5)) * 32;
    int  nt0  = blockIdx.y * 4;
    v8f c[2][4] = {};
    gemm_core(A, 256, W, 48, 8, row0, nt0, c);
    const float scale = 0.17677669529663689f;   // 32^-0.5
#pragma unroll
    for (int mt = 0; mt < 2; ++mt)
#pragma unroll
        for (int nt = 0; nt < 4; ++nt) {
            int n = (nt0 + nt) * 16 + ncol;
            float b = bias[n];
            float sc = (n < 256) ? scale : 1.0f;
#pragma unroll
            for (int j = 0; j < 8; ++j) {
                long r = row0 + mt * 16 + j + g * 8;
                out[r * 768 + n] = f2bf((c[mt][nt][j] + b) * sc);
            }
        }
}

// --- GEMM 2: proj + residual; window-order rows scattered to token-major xc2
__global__ __launch_bounds__(256) void k_gemm_proj(const unsigned short* __restrict__ A,
                                                   const unsigned short* __restrict__ W,
                                                   const float* __restrict__ bias,
                                                   const float* __restrict__ xc,
                                                   float* __restrict__ xc2) {
    const int lane = threadIdx.x & 31, g = lane >> 4, ncol = lane & 15;
    long row0 = ((long)blockIdx.x * 8 + (threadIdx.x >> 5)) * 32;
    int  nt0  = blockIdx.y * 4;
    v8f c[2][4] = {};
    gemm_core(A, 256, W, 16, 8, row0, nt0, c);
#pragma unroll
    for (int mt = 0; mt < 2; ++mt)
#pragma unroll
        for (int j = 0; j < 8; ++j) {
            long r   = row0 + mt * 16 + j + g * 8;      // window-ordered row
            int  win = (int)(r >> 6), t = (int)r & 63;
            int  b = win >> 8, wy = (win >> 4) & 15, wx = win & 15;
            int  hh = wy * 8 + (t >> 3), ww = wx * 8 + (t & 7);
            long tok = ((long)b << 14) + (hh << 7) + ww;
#pragma unroll
            for (int nt = 0; nt < 4; ++nt) {
                int  n   = (nt0 + nt) * 16 + ncol;
                long idx = tok * NC + n;
                xc2[idx] = xc[idx] + c[mt][nt][j] + bias[n];
            }
        }
}

// --- GEMM 3: mlp1 + exact GELU, bf16 out
__global__ __launch_bounds__(256) void k_gemm_mlp1(const unsigned short* __restrict__ A,
                                                   const unsigned short* __restrict__ W,
                                                   const float* __restrict__ bias,
                                                   unsigned short* __restrict__ out) {
    const int lane = threadIdx.x & 31, g = lane >> 4, ncol = lane & 15;
    long row0 = ((long)blockIdx.x * 8 + (threadIdx.x >> 5)) * 32;
    int  nt0  = blockIdx.y * 4;
    v8f c[2][4] = {};
    gemm_core(A, 256, W, 64, 8, row0, nt0, c);
#pragma unroll
    for (int mt = 0; mt < 2; ++mt)
#pragma unroll
        for (int nt = 0; nt < 4; ++nt) {
            int n = (nt0 + nt) * 16 + ncol;
            float b = bias[n];
#pragma unroll
            for (int j = 0; j < 8; ++j) {
                long r = row0 + mt * 16 + j + g * 8;
                float v = c[mt][nt][j] + b;
                float gl = 0.5f * v * (1.0f + erff(v * 0.70710678118654752f));
                out[r * 1024 + n] = f2bf(gl);
            }
        }
}

// --- GEMM 4: mlp2 + residual + NHWC->NCHW scatter to d_out (fp32)
__global__ __launch_bounds__(256) void k_gemm_mlp2(const unsigned short* __restrict__ A,
                                                   const unsigned short* __restrict__ W,
                                                   const float* __restrict__ bias,
                                                   const float* __restrict__ xc2,
                                                   float* __restrict__ out) {
    const int lane = threadIdx.x & 31, g = lane >> 4, ncol = lane & 15;
    long row0 = ((long)blockIdx.x * 8 + (threadIdx.x >> 5)) * 32;
    int  nt0  = blockIdx.y * 4;
    v8f c[2][4] = {};
    gemm_core(A, 1024, W, 16, 32, row0, nt0, c);
#pragma unroll
    for (int mt = 0; mt < 2; ++mt)
#pragma unroll
        for (int j = 0; j < 8; ++j) {
            long r  = row0 + mt * 16 + j + g * 8;     // token-major row
            long b  = r >> 14;
            long hw = r & 16383;
#pragma unroll
            for (int nt = 0; nt < 4; ++nt) {
                int n = (nt0 + nt) * 16 + ncol;
                out[(b * NC + n) * 16384 + hw] =
                    xc2[r * NC + n] + c[mt][nt][j] + bias[n];
            }
        }
}

// ---------------------------------------------------------------------------
// Fused window attention: one workgroup per window, one wave per head.
// S = (q*scale)·k^T + rel_bias ; softmax ; out = P·v.  Per-wave LDS region:
// P (64x64 bf16, 8KB) + vT (32x64 bf16, 4KB) = 12KB; 8 waves -> 96KB dynamic.
// ---------------------------------------------------------------------------
__global__ __launch_bounds__(256) void k_attention(const unsigned short* __restrict__ qkv,
                                                   const float* __restrict__ relt,
                                                   unsigned short* __restrict__ attn_out) {
    extern __shared__ unsigned short sm[];
    const int win  = blockIdx.x;
    const int wid  = threadIdx.x >> 5;      // = head
    const int lane = threadIdx.x & 31;
    const int g = lane >> 4, ncol = lane & 15;
    const int hd = wid;
    unsigned short* P  = sm + wid * 6144;
    unsigned short* vT = P + 4096;
    const long base = (long)win * 64 * 768;

    // --- stage v transposed into LDS: vT[d][token], d = lane
    for (int t = 0; t < 64; t += 2) {
        unsigned u0 = qkv[base + (long)t       * 768 + 512 + hd * 32 + lane];
        unsigned u1 = qkv[base + (long)(t + 1) * 768 + 512 + hd * 32 + lane];
        *(unsigned*)&vT[lane * 64 + t] = u0 | (u1 << 16);
    }

    // --- load q as 4 A-tiles (16x32, K=d) and k as 4 B-tiles (32x16)
    BV qa[4], kb[4];
#pragma unroll
    for (int tr = 0; tr < 4; ++tr) {
        const unsigned short* p =
            qkv + base + (long)(tr * 16 + ncol) * 768 + hd * 32 + g * 16;
        qa[tr].u4[0] = *(const v4u*)p;   qa[tr].u4[1] = *(const v4u*)(p + 8);
    }
#pragma unroll
    for (int tn = 0; tn < 4; ++tn) {
        const unsigned short* p =
            qkv + base + (long)(tn * 16 + ncol) * 768 + 256 + hd * 32 + g * 16;
        kb[tn].u4[0] = *(const v4u*)p;   kb[tn].u4[1] = *(const v4u*)(p + 8);
    }

    // --- S = q·k^T  (16 WMMAs, K=32=d)
    v8f S[4][4] = {};
#pragma unroll
    for (int tr = 0; tr < 4; ++tr)
#pragma unroll
        for (int tn = 0; tn < 4; ++tn)
            S[tr][tn] = wmma_bf16(qa[tr].bf, kb[tn].bf, S[tr][tn]);

    // --- bias + softmax (rows live in 16-lane groups), write P bf16 to LDS
#pragma unroll
    for (int tr = 0; tr < 4; ++tr) {
#pragma unroll
        for (int j = 0; j < 8; ++j) {
            int r  = tr * 16 + j + g * 8;
            int ry = r >> 3, rx = r & 7;
            float v[4];
#pragma unroll
            for (int tc = 0; tc < 4; ++tc) {
                int cidx = tc * 16 + ncol;
                int bi = (ry - (cidx >> 3) + 7) * 15 + (rx - (cidx & 7) + 7);
                v[tc] = S[tr][tc][j] + relt[bi * 8 + hd];
            }
            float m = fmaxf(fmaxf(v[0], v[1]), fmaxf(v[2], v[3]));
#pragma unroll
            for (int msk = 1; msk < 16; msk <<= 1)
                m = fmaxf(m, __shfl_xor(m, msk, 32));
            float e[4], s = 0.f;
#pragma unroll
            for (int tc = 0; tc < 4; ++tc) { e[tc] = __expf(v[tc] - m); s += e[tc]; }
#pragma unroll
            for (int msk = 1; msk < 16; msk <<= 1)
                s += __shfl_xor(s, msk, 32);
            float inv = 1.0f / s;
#pragma unroll
            for (int tc = 0; tc < 4; ++tc)
                P[r * 64 + tc * 16 + ncol] = f2bf(e[tc] * inv);
        }
    }

    // --- out = P(64x64) · v(64x32): 8 tiles, K=64 (2 WMMAs each)
#pragma unroll
    for (int otr = 0; otr < 4; ++otr)
#pragma unroll
        for (int otc = 0; otc < 2; ++otc) {
            v8f acc = {};
#pragma unroll
            for (int k2 = 0; k2 < 2; ++k2) {
                BV pa, vb;
                const unsigned short* pp =
                    P + (otr * 16 + ncol) * 64 + k2 * 32 + g * 16;
                pa.u4[0] = *(const v4u*)pp;  pa.u4[1] = *(const v4u*)(pp + 8);
                const unsigned short* vp =
                    vT + (otc * 16 + ncol) * 64 + k2 * 32 + g * 16;
                vb.u4[0] = *(const v4u*)vp;  vb.u4[1] = *(const v4u*)(vp + 8);
                acc = wmma_bf16(pa.bf, vb.bf, acc);
            }
#pragma unroll
            for (int j = 0; j < 8; ++j) {
                int tok = otr * 16 + j + g * 8;
                int dch = otc * 16 + ncol;
                attn_out[((long)win * 64 + tok) * NC + hd * 32 + dch] = f2bf(acc[j]);
            }
        }
}

// ---------------------------------------------------------------------------
// Host launcher
// ---------------------------------------------------------------------------
extern "C" void kernel_launch(void* const* d_in, const int* in_sizes, int n_in,
                              void* d_out, int out_size, void* d_ws, size_t ws_size,
                              hipStream_t stream) {
    (void)in_sizes; (void)n_in; (void)out_size; (void)ws_size;
    const float* x       = (const float*)d_in[0];
    const float* ln1_s   = (const float*)d_in[1];
    const float* ln1_b   = (const float*)d_in[2];
    const float* w_qkv   = (const float*)d_in[3];
    const float* b_qkv   = (const float*)d_in[4];
    const float* w_proj  = (const float*)d_in[5];
    const float* b_proj  = (const float*)d_in[6];
    const float* rel_tab = (const float*)d_in[7];
    const float* ln2_s   = (const float*)d_in[8];
    const float* ln2_b   = (const float*)d_in[9];
    const float* w_mlp1  = (const float*)d_in[10];
    const float* b_mlp1  = (const float*)d_in[11];
    const float* w_mlp2  = (const float*)d_in[12];
    const float* b_mlp2  = (const float*)d_in[13];
    float* out = (float*)d_out;

    char* ws = (char*)d_ws;
    // workspace layout (bytes); G (MLP hidden) aliases dead XC+XN+ATT region
    float*          xc   = (float*)(ws + 0L);                    // 134217728
    unsigned short* xn   = (unsigned short*)(ws + 134217728L);   //  67108864
    unsigned short* att  = (unsigned short*)(ws + 201326592L);   //  67108864
    unsigned short* qkv  = (unsigned short*)(ws + 268435456L);   // 201326592
    float*          xc2  = (float*)(ws + 469762048L);            // 134217728
    unsigned short* h2   = (unsigned short*)(ws + 603979776L);   //  67108864
    unsigned short* wq   = (unsigned short*)(ws + 671088640L);   //    393216
    unsigned short* wp   = (unsigned short*)(ws + 671481856L);   //    131072
    unsigned short* w1   = (unsigned short*)(ws + 671612928L);   //    524288
    unsigned short* w2   = (unsigned short*)(ws + 672137216L);   //    524288
    unsigned short* gbuf = (unsigned short*)(ws + 0L);           // 268435456 (alias)

    // 0) NCHW -> NHWC
    k_transpose<<<dim3(512, 8, 8), dim3(32, 8), 0, stream>>>(x, xc);

    // weight pre-swizzles (tiny)
    k_prep_w<<<48, 256, 0, stream>>>(w_qkv, wq, 256, 768);
    k_prep_w<<<16, 256, 0, stream>>>(w_proj, wp, 256, 256);
    k_prep_w<<<64, 256, 0, stream>>>(w_mlp1, w1, 256, 1024);
    k_prep_w<<<64, 256, 0, stream>>>(w_mlp2, w2, 1024, 256);

    // 1) LN1 -> window-partitioned bf16
    k_layernorm<<<16384, 256, 0, stream>>>(xc, ln1_s, ln1_b, xn, 1);

    // 2) qkv GEMM (M=131072, N=768, K=256), q pre-scaled
    k_gemm_qkv<<<dim3(512, 12), 256, 0, stream>>>(xn, wq, b_qkv, qkv);

    // 3) fused window attention (2048 windows, 8 heads, 96KB dyn LDS)
    k_attention<<<2048, 256, 98304, stream>>>(qkv, rel_tab, att);

    // 4) proj GEMM + residual -> xc2 (token-major)
    k_gemm_proj<<<dim3(512, 4), 256, 0, stream>>>(att, wp, b_proj, xc, xc2);

    // 5) LN2 -> bf16
    k_layernorm<<<16384, 256, 0, stream>>>(xc2, ln2_s, ln2_b, h2, 0);

    // 6) MLP1 + GELU (N=1024)
    k_gemm_mlp1<<<dim3(512, 16), 256, 0, stream>>>(h2, w1, b_mlp1, gbuf);

    // 7) MLP2 + residual + NCHW scatter (K=1024)
    k_gemm_mlp2<<<dim3(512, 4), 256, 0, stream>>>(gbuf, w2, b_mlp2, xc2, out);
}
